// DementiaModel_37769942401222
// MI455X (gfx1250) — matrix operand order repaired and checked by reading, not compiled, not run on hardware
//
#include <hip/hip_runtime.h>

// ---------------------------------------------------------------------------
// Fused 2-layer LSTM (H=64, D=8, T=256) + FC(64->1) + sigmoid for MI455X.
//
// A 4-wave workgroup owns 16 batch rows jointly. Wave w owns hidden-column
// block j=w, i.e. gate tiles {w, w+4, w+8, w+12} (its own i/f/g/o columns),
// so its 28 WMMA B-fragments (f16 weights) live PERMANENTLY in VGPRs
// (~224 regs) -- zero per-step weight traffic. Per-step h exchange
// (C-layout -> A-layout transpose + cross-wave broadcast) goes through 4KB
// of LDS with two __syncthreads per step. Accumulators start from inline-0
// C operands; biases are added in the activation phase. Activations are
// branch-free native TRANS ops (v_tanh_f32 when exposed, else exp2-based;
// sigmoid via v_exp_f32 + v_rcp_f32).
//
// Round-4 polish: constant-offset LDS staging stores (fold into ds_store
// offset field, killing per-step address adds), near-cache prefetch
// locality for the streaming x reads, exact 2*log2e constant, and a wider
// native-tanh builtin probe.
// ---------------------------------------------------------------------------

#define HID    64
#define GATES  256   // 4*HID
#define DIN    8
#define TSTEPS 256
#define WAVES  4     // one wave per 16-wide hidden-column block
#define RPW    16    // batch rows per workgroup

typedef __attribute__((ext_vector_type(16))) _Float16 v16h;
typedef __attribute__((ext_vector_type(8)))  _Float16 v8h;
typedef __attribute__((ext_vector_type(8)))  float    v8f;
typedef __attribute__((ext_vector_type(4)))  float    f32x4;

union U16 { v16h v; v8h h[2]; };

#define LOG2E  1.4426950408889634f
#define LOG2E2 2.8853900817779268f

// sigmoid(x) = 1 / (1 + 2^(-x*log2e)); v_exp_f32 is exp2 on AMD.
// x -> -inf: exp2 -> +inf -> rcp -> 0.  x -> +inf: exp2 -> 0 -> 1.  Branch-free.
__device__ inline float fast_sigm(float v) {
  return __builtin_amdgcn_rcpf(1.f + __builtin_amdgcn_exp2f(-LOG2E * v));
}

__device__ inline float fast_tanh(float v) {
#if __has_builtin(__builtin_amdgcn_tanhf)
  return __builtin_amdgcn_tanhf(v);            // native v_tanh_f32 (CDNA5)
#elif __has_builtin(__builtin_amdgcn_tanh_f32)
  return __builtin_amdgcn_tanh_f32(v);
#else
  // tanh(x) = 1 - 2/(1 + 2^(2x*log2e)); saturates correctly, branch-free.
  const float e = __builtin_amdgcn_exp2f(LOG2E2 * v);
  return 1.f - 2.f * __builtin_amdgcn_rcpf(1.f + e);
#endif
}

#define WMMA(a, b, c) __builtin_amdgcn_wmma_f32_16x16x32_f16( \
    false, (a), false, (b), (short)0, (c), false, false)

// Build a B fragment (contiguous-K order, same as ldsA) in registers from a
// global row-major weight [GATES][Kdim] (fp32 -> f16). Lane l holds column
// tile*16+(l&15); lanes 0-15 carry K=kc..kc+15, lanes 16-31 K=kc+16..kc+31.
// k >= Kdim is zero-padded (pads Wih0's K from 8 to 32).
__device__ inline v16h gBfrag(const float* __restrict__ W, int Kdim,
                              int tile, int kc, int lane) {
  const int col = tile * 16 + (lane & 15);
  const int ks  = kc + ((lane & 16) ? 16 : 0);
  v16h r;
  #pragma unroll
  for (int i = 0; i < 16; ++i) {
    const int k = ks + i;
    r[i] = (_Float16)((k < Kdim) ? W[col * Kdim + k] : 0.f);
  }
  return r;
}

// A fragments (K=0..31 and K=32..63) from a [16][64] f16 row-major LDS
// staging buffer. Lane l = row (l&15); lanes 0-15 take K {0..15}(+kc),
// lanes 16-31 take K {16..31}(+kc).
__device__ inline void ldsA(const _Float16* hst, int lane, v16h& lo, v16h& hi) {
  const int m  = lane & 15;
  const int kb = (lane & 16) ? 16 : 0;
  const _Float16* p = hst + m * HID;
  U16 u;
  u.h[0] = *(const v8h*)(p + kb);
  u.h[1] = *(const v8h*)(p + 8 + kb);
  lo = u.v;
  u.h[0] = *(const v8h*)(p + 32 + kb);
  u.h[1] = *(const v8h*)(p + 40 + kb);
  hi = u.v;
}

__global__ __launch_bounds__(WAVES * 32, 1)
void lstm2_fc_kernel(const float* __restrict__ x,
                     const float* __restrict__ Wih0, const float* __restrict__ Whh0,
                     const float* __restrict__ bih0, const float* __restrict__ bhh0,
                     const float* __restrict__ Wih1, const float* __restrict__ Whh1,
                     const float* __restrict__ bih1, const float* __restrict__ bhh1,
                     const float* __restrict__ Wfc,  const float* __restrict__ bfc,
                     float* __restrict__ out)
{
  __shared__ _Float16 sH0 [RPW * HID];   // layer-0 h exchange ([row][k], f16)
  __shared__ _Float16 sH1 [RPW * HID];   // layer-1 h exchange
  __shared__ float    sFin[RPW * HID];   // last-step layer-1 h (fp32, FC tail)

  const int tid   = threadIdx.x;
  const int wave  = tid >> 5;
  const int lane  = tid & 31;
  const int rowBase = blockIdx.x * RPW;
  const int nloc  = lane & 15;           // column within this wave's 16-block
  const int ncol  = wave * 16 + nloc;    // hidden column owned (0..63)
  const int mbase = (lane & 16) ? 8 : 0; // C-layout row base for this lane

  // Per-lane staging bases: element r lands at constant byte offset r*HID*2,
  // which folds into the ds_store immediate offset field.
  _Float16* st0p = sH0  + mbase * HID + ncol;
  _Float16* st1p = sH1  + mbase * HID + ncol;
  float*    finp = sFin + mbase * HID + ncol;

  // ---- persistent register-resident weight fragments for this wave ----
  v16h fW0 [4];                          // Wih0 tiles, K padded 8->32 (1 chunk)
  v16h fWh0[4][2], fW1[4][2], fWh1[4][2];
  float b0[4], b1[4];                    // per-lane gate biases (i,f,g,o)
  #pragma unroll
  for (int q = 0; q < 4; ++q) {
    const int tile = wave + 4 * q;       // q: 0=i 1=f 2=g 3=o
    fW0 [q]    = gBfrag(Wih0, DIN, tile, 0,  lane);
    fWh0[q][0] = gBfrag(Whh0, HID, tile, 0,  lane);
    fWh0[q][1] = gBfrag(Whh0, HID, tile, 32, lane);
    fW1 [q][0] = gBfrag(Wih1, HID, tile, 0,  lane);
    fW1 [q][1] = gBfrag(Wih1, HID, tile, 32, lane);
    fWh1[q][0] = gBfrag(Whh1, HID, tile, 0,  lane);
    fWh1[q][1] = gBfrag(Whh1, HID, tile, 32, lane);
    const int g = tile * 16 + nloc;
    b0[q] = bih0[g] + bhh0[g];
    b1[q] = bih1[g] + bhh1[g];
  }

  // ---- persistent state ----
  v8f c0 = (v8f)0.f, c1 = (v8f)0.f;      // this wave's 16x16 c blocks (C-layout)
  v16h h0lo, h0hi, h1lo, h1hi;           // full-H A fragments (shared via LDS)
  #pragma unroll
  for (int i = 0; i < 16; ++i) {
    h0lo[i] = (_Float16)0.f; h0hi[i] = (_Float16)0.f;
    h1lo[i] = (_Float16)0.f; h1hi[i] = (_Float16)0.f;
  }

  const float* xrow = x + (size_t)(rowBase + nloc) * (TSTEPS * DIN);

  for (int t = 0; t < TSTEPS; ++t) {
    // ---- x A-fragment (K padded 8->32; only lanes 0..15 carry data) ----
    v16h xa;
    #pragma unroll
    for (int i = 0; i < 16; ++i) xa[i] = (_Float16)0.f;
    if (lane < 16) {
      const f32x4* xp = (const f32x4*)(xrow + t * DIN);
      const f32x4 a0 = xp[0], a1 = xp[1];
      xa[0] = (_Float16)a0.x; xa[1] = (_Float16)a0.y;
      xa[2] = (_Float16)a0.z; xa[3] = (_Float16)a0.w;
      xa[4] = (_Float16)a1.x; xa[5] = (_Float16)a1.y;
      xa[6] = (_Float16)a1.z; xa[7] = (_Float16)a1.w;
      if (t + 1 < TSTEPS) __builtin_prefetch(xrow + (t + 1) * DIN, 0, 3);
    }

    v8f acc[4];

    // -------- layer 0: this wave's i/f/g/o tiles (12 WMMA) --------
    #pragma unroll
    for (int q = 0; q < 4; ++q) {
      acc[q] = WMMA(xa,   fW0 [q],    (v8f)0.f);   // C = inline 0
      acc[q] = WMMA(h0lo, fWh0[q][0], acc[q]);
      acc[q] = WMMA(h0hi, fWh0[q][1], acc[q]);
    }
    #pragma unroll
    for (int r = 0; r < 8; ++r) {
      const float ig = fast_sigm(acc[0][r] + b0[0]);
      const float fg = fast_sigm(acc[1][r] + b0[1]);
      const float gg = fast_tanh(acc[2][r] + b0[2]);
      const float og = fast_sigm(acc[3][r] + b0[3]);
      const float c  = fg * c0[r] + ig * gg;
      c0[r] = c;
      st0p[r * HID] = (_Float16)(og * fast_tanh(c));
    }
    __syncthreads();                      // drains DS, publishes h0(t)
    ldsA(sH0, lane, h0lo, h0hi);          // full-H h0 A-fragments

    // -------- layer 1: gates = h0@Wih1^T + h1@Whh1^T (16 WMMA) --------
    #pragma unroll
    for (int q = 0; q < 4; ++q) {
      acc[q] = WMMA(h0lo, fW1 [q][0], (v8f)0.f);
      acc[q] = WMMA(h0hi, fW1 [q][1], acc[q]);
      acc[q] = WMMA(h1lo, fWh1[q][0], acc[q]);
      acc[q] = WMMA(h1hi, fWh1[q][1], acc[q]);
    }
    #pragma unroll
    for (int r = 0; r < 8; ++r) {
      const float ig = fast_sigm(acc[0][r] + b1[0]);
      const float fg = fast_sigm(acc[1][r] + b1[1]);
      const float gg = fast_tanh(acc[2][r] + b1[2]);
      const float og = fast_sigm(acc[3][r] + b1[3]);
      const float c  = fg * c1[r] + ig * gg;
      c1[r] = c;
      const float h = og * fast_tanh(c);
      st1p[r * HID] = (_Float16)h;
      if (t == TSTEPS - 1) finp[r * HID] = h;
    }
    __syncthreads();                      // publishes h1(t) (and sFin last step)
    ldsA(sH1, lane, h1lo, h1hi);          // h1 A-fragments for step t+1
  }

  // -------- FC tail: out = sigmoid(h1_last . Wfc + bfc) --------
  if (wave == 0 && lane < 16) {
    const float* fr = sFin + lane * HID;
    float acc = bfc[0];
    #pragma unroll
    for (int k = 0; k < HID; ++k) acc += fr[k] * Wfc[k];
    out[rowBase + lane] = fast_sigm(acc);
  }
}

extern "C" void kernel_launch(void* const* d_in, const int* in_sizes, int n_in,
                              void* d_out, int out_size, void* d_ws, size_t ws_size,
                              hipStream_t stream) {
  const float* x    = (const float*)d_in[0];
  const float* Wih0 = (const float*)d_in[1];
  const float* Whh0 = (const float*)d_in[2];
  const float* bih0 = (const float*)d_in[3];
  const float* bhh0 = (const float*)d_in[4];
  const float* Wih1 = (const float*)d_in[5];
  const float* Whh1 = (const float*)d_in[6];
  const float* bih1 = (const float*)d_in[7];
  const float* bhh1 = (const float*)d_in[8];
  const float* Wfc  = (const float*)d_in[9];
  const float* bfc  = (const float*)d_in[10];
  float* out = (float*)d_out;

  const int B    = in_sizes[0] / (TSTEPS * DIN);   // 8192
  const int grid = B / RPW;                        // 512 workgroups x 4 waves

  lstm2_fc_kernel<<<grid, WAVES * 32, 0, stream>>>(
      x, Wih0, Whh0, bih0, bhh0, Wih1, Whh1, bih1, bhh1, Wfc, bfc, out);
}
